// SLAYERDense_63213328662984
// MI455X (gfx1250) — matrix operand order repaired and checked by reading, not compiled, not run on hardware
//
#include <hip/hip_runtime.h>
#include <hip/hip_bf16.h>

// ---------------------------------------------------------------------------
// SLAYER CUBA 3-layer SNN forward for MI455X (gfx1250, wave32, WMMA).
//
// Pipeline:
//   1) convert W1/W2/W3 fp32 -> fp16
//   2) transpose spike [B, N_IN, T] fp32 -> S0h [T*B, N_IN] fp16 (LDS tiled)
//   3) WMMA GEMM (double-buffered async-to-LDS): Z[n,o] = sum_k S[n,k]*W[o,k]
//   4) scan over T -> S1h fp16 spikes
//   5) WMMA GEMM with W2h -> Z ; scan -> S2h
//   6) tiny layer-3 dot products (O=2) + scan -> d_out [B,2,T] fp32
// ---------------------------------------------------------------------------

#define BB   64
#define NIN  2048
#define HH   1024
#define NOUT 2
#define TT   128
#define NN   (BB * TT)        // 8192 GEMM "N" dimension (t*B + b)

typedef _Float16 v16h __attribute__((ext_vector_type(16)));
typedef _Float16 v8h  __attribute__((ext_vector_type(8)));
typedef float    v8f  __attribute__((ext_vector_type(8)));
typedef unsigned long long u64x2 __attribute__((ext_vector_type(2)));
typedef int v4i __attribute__((vector_size(16)));   // matches async-LDS builtin params

union Frag16 { v16h v; v8h h[2]; };

#if defined(__has_builtin)
#if __has_builtin(__builtin_amdgcn_global_load_async_to_lds_b128) && \
    __has_builtin(__builtin_amdgcn_s_wait_asynccnt)
#define HAVE_ASYNC_LDS 1
#endif
#endif

#define GAS __attribute__((address_space(1)))
#define LAS __attribute__((address_space(3)))

// ---------------- fp32 -> fp16 convert ----------------
__global__ void cvt_f32_to_f16(const float* __restrict__ src,
                               _Float16* __restrict__ dst, int n) {
    int i = blockIdx.x * blockDim.x + threadIdx.x;
    if (i < n) dst[i] = (_Float16)src[i];
}

// ---------------- spike transpose [B,NIN,T] -> [T*B, NIN] fp16 -------------
// grid: (T/32, NIN/32, B), block: (32, 8)
__global__ void spike_transpose(const float* __restrict__ spike,
                                _Float16* __restrict__ out) {
    __shared__ float tile[32][33];
    const int t0 = blockIdx.x * 32;
    const int i0 = blockIdx.y * 32;
    const int b  = blockIdx.z;
#pragma unroll
    for (int r = 0; r < 4; ++r) {
        int il = threadIdx.y + r * 8;
        tile[il][threadIdx.x] =
            spike[((size_t)b * NIN + (i0 + il)) * TT + (t0 + threadIdx.x)];
    }
    __syncthreads();
#pragma unroll
    for (int r = 0; r < 4; ++r) {
        int tl = threadIdx.y + r * 8;
        out[((size_t)(t0 + tl) * BB + b) * NIN + i0 + threadIdx.x] =
            (_Float16)tile[threadIdx.x][tl];
    }
}

// ---------------- WMMA GEMM:  Z[n,o] = sum_k Sh[n,k] * Wh[o,k] -------------
// A = S tile (M = n), B = W tile (N = o)  -> D lanes map to consecutive o,
// so epilogue stores are 64B-contiguous per half-wave.
// Workgroup tile: 128 n x 64 o, 256 threads = 8 waves (4 n-dir x 2 o-dir).
// Each wave: 2x2 sub-tiles of 16x16, K step 32, double-buffered LDS with
// async global->LDS loads (ASYNCcnt) when available.
__global__ __launch_bounds__(256)
void wmma_gemm(const _Float16* __restrict__ Wh,  // [O, K] row-major fp16
               const _Float16* __restrict__ Sh,  // [N, K] row-major fp16
               float* __restrict__ Z,            // [N, O] row-major fp32
               int O, int K) {
    __shared__ alignas(16) _Float16 Ws[2][64 * 32];   // [buf][o][k]
    __shared__ alignas(16) _Float16 Ss[2][128 * 32];  // [buf][n][k]

    const int tid   = threadIdx.x;
    const int lane  = tid & 31;
    const int wave  = tid >> 5;
    const int waveN = wave & 3;   // 0..3  (n direction)
    const int waveO = wave >> 2;  // 0..1  (o direction)
    const int wgO   = blockIdx.x * 64;
    const int wgN   = blockIdx.y * 128;
    const int laneM = lane & 15;
    const int hi    = lane >> 4;  // 0 or 1

    const v8f vzero = {0.f, 0.f, 0.f, 0.f, 0.f, 0.f, 0.f, 0.f};
    v8f acc[2][2];                // [i: n-subtile][j: o-subtile]
    acc[0][0] = vzero; acc[0][1] = vzero; acc[1][0] = vzero; acc[1][1] = vzero;

    // per-thread staging assignments
    const int wrow = tid >> 2;            // 0..63   (o row in W tile)
    const int wcol = (tid & 3) << 3;      // 0,8,16,24
    const int srow = tid >> 1;            // 0..127  (n row in S tile)
    const int scol = (tid & 1) << 4;      // 0 or 16

    const _Float16* gW = Wh + (size_t)(wgO + wrow) * K + wcol;
    const _Float16* gS = Sh + (size_t)(wgN + srow) * K + scol;

#if defined(HAVE_ASYNC_LDS)
    // ---- async double-buffered pipeline --------------------------------
    auto stage = [&](int buf, int kk) {
        __builtin_amdgcn_global_load_async_to_lds_b128(
            (GAS v4i*)(gW + kk), (LAS v4i*)(&Ws[buf][wrow * 32 + wcol]), 0, 0);
        __builtin_amdgcn_global_load_async_to_lds_b128(
            (GAS v4i*)(gS + kk), (LAS v4i*)(&Ss[buf][srow * 32 + scol]), 0, 0);
        __builtin_amdgcn_global_load_async_to_lds_b128(
            (GAS v4i*)(gS + kk + 8), (LAS v4i*)(&Ss[buf][srow * 32 + scol + 8]), 0, 0);
    };
    stage(0, 0);
    __builtin_amdgcn_s_wait_asynccnt(0);
    __syncthreads();

    for (int kk = 0; kk < K; kk += 32) {
        const int cur  = (kk >> 5) & 1;
        const bool more = (kk + 32) < K;
        if (more) {
            stage(1 - cur, kk + 32);
            if (kk + 64 < K) {   // prefetch the tile after next into L2
                __builtin_prefetch(gW + kk + 64, 0, 1);
                __builtin_prefetch(gS + kk + 64, 0, 1);
            }
        }
        // A fragments (S): lane = row M=n, K chunks {k0..k0+7},{k0+16..k0+23}
        Frag16 aS[2];
        const int k0 = hi ? 8 : 0;
#pragma unroll
        for (int i = 0; i < 2; ++i) {
            int M = waveN * 32 + i * 16 + laneM;
            aS[i].h[0] = *(const v8h*)(&Ss[cur][M * 32 + k0]);
            aS[i].h[1] = *(const v8h*)(&Ss[cur][M * 32 + k0 + 16]);
        }
        // B fragments (W): lane = column N=o, K chunks {c0..c0+7},{c0+16..c0+23}
        Frag16 bW[2];
        const int c0 = hi ? 8 : 0;
#pragma unroll
        for (int j = 0; j < 2; ++j) {
            int Nc = waveO * 32 + j * 16 + laneM;
            bW[j].h[0] = *(const v8h*)(&Ws[cur][Nc * 32 + c0]);
            bW[j].h[1] = *(const v8h*)(&Ws[cur][Nc * 32 + c0 + 16]);
        }
#pragma unroll
        for (int i = 0; i < 2; ++i)
#pragma unroll
            for (int j = 0; j < 2; ++j)
                acc[i][j] = __builtin_amdgcn_wmma_f32_16x16x32_f16(
                    false, aS[i].v, false, bW[j].v,
                    (short)0, acc[i][j], false, false);
        if (more) __builtin_amdgcn_s_wait_asynccnt(0);
        __syncthreads();
    }
#else
    // ---- fallback: synchronous staging (single buffer) ------------------
    for (int kk = 0; kk < K; kk += 32) {
        *(u64x2*)(&Ws[0][wrow * 32 + wcol])     = *(const u64x2*)(gW + kk);
        *(u64x2*)(&Ss[0][srow * 32 + scol])     = *(const u64x2*)(gS + kk);
        *(u64x2*)(&Ss[0][srow * 32 + scol + 8]) = *(const u64x2*)(gS + kk + 8);
        if (kk + 32 < K) {
            __builtin_prefetch(gW + kk + 32, 0, 1);
            __builtin_prefetch(gS + kk + 32, 0, 1);
        }
        __syncthreads();
        Frag16 aS[2];
        const int k0 = hi ? 8 : 0;
#pragma unroll
        for (int i = 0; i < 2; ++i) {
            int M = waveN * 32 + i * 16 + laneM;
            aS[i].h[0] = *(const v8h*)(&Ss[0][M * 32 + k0]);
            aS[i].h[1] = *(const v8h*)(&Ss[0][M * 32 + k0 + 16]);
        }
        Frag16 bW[2];
        const int c0 = hi ? 8 : 0;
#pragma unroll
        for (int j = 0; j < 2; ++j) {
            int Nc = waveO * 32 + j * 16 + laneM;
            bW[j].h[0] = *(const v8h*)(&Ws[0][Nc * 32 + c0]);
            bW[j].h[1] = *(const v8h*)(&Ws[0][Nc * 32 + c0 + 16]);
        }
#pragma unroll
        for (int i = 0; i < 2; ++i)
#pragma unroll
            for (int j = 0; j < 2; ++j)
                acc[i][j] = __builtin_amdgcn_wmma_f32_16x16x32_f16(
                    false, aS[i].v, false, bW[j].v,
                    (short)0, acc[i][j], false, false);
        __syncthreads();
    }
#endif

    // epilogue: D lanes 0-15 = N (o, consecutive) ; VGPR v -> M=n (+8 hi half)
    const int nbase = wgN + waveN * 32;
    const int obase = wgO + waveO * 32;
#pragma unroll
    for (int j = 0; j < 2; ++j) {
        int o = obase + j * 16 + laneM;
#pragma unroll
        for (int i = 0; i < 2; ++i) {
#pragma unroll
            for (int v = 0; v < 8; ++v) {
                int n = nbase + i * 16 + v + (hi ? 8 : 0);
                Z[(size_t)n * O + o] = acc[i][j][v];
            }
        }
    }
}

// ---------------- CUBA leaky-IF scan over T, fp16 spike output -------------
__global__ void cuba_scan(const float* __restrict__ Z,
                          _Float16* __restrict__ Sout, int BO) {
    int id = blockIdx.x * blockDim.x + threadIdx.x;
    if (id >= BO) return;
    float cur = 0.f, vol = 0.f;
    for (int t = 0; t < TT; ++t) {
        float z = Z[(size_t)t * BO + id];
        cur = cur * 0.7f + z;           // 1 - CURRENT_DECAY(0.3)
        vol = vol * 0.2f + cur;         // 1 - VOLTAGE_DECAY(0.8)
        float s = (vol >= 1.0f) ? 1.0f : 0.0f;   // threshold spike
        vol *= (1.0f - s);              // reset on spike
        Sout[(size_t)t * BO + id] = (_Float16)s;
    }
}

// ---------------- layer 3: z3[n, o] = sum_h W3h[o,h] * S2h[n,h], O=2 -------
__global__ void layer3_z(const _Float16* __restrict__ W3h,
                         const _Float16* __restrict__ S2h,
                         float* __restrict__ Z3) {
    const int lane = threadIdx.x & 31;
    const int wave = threadIdx.x >> 5;
    const int n = blockIdx.x * 8 + wave;
    float a0 = 0.f, a1 = 0.f;
    for (int h = lane; h < HH; h += 32) {
        float s = (float)S2h[(size_t)n * HH + h];
        a0 += s * (float)W3h[h];
        a1 += s * (float)W3h[HH + h];
    }
#pragma unroll
    for (int off = 16; off > 0; off >>= 1) {
        a0 += __shfl_xor(a0, off, 32);
        a1 += __shfl_xor(a1, off, 32);
    }
    if (lane == 0) {
        Z3[(size_t)n * 2 + 0] = a0;
        Z3[(size_t)n * 2 + 1] = a1;
    }
}

// ---------------- final scan -> d_out [B, 2, T] fp32 -----------------------
__global__ void final_scan(const float* __restrict__ Z3,
                           float* __restrict__ out) {
    int id = threadIdx.x;            // 0..127 = B*NOUT
    int b = id >> 1, o = id & 1;
    float cur = 0.f, vol = 0.f;
    for (int t = 0; t < TT; ++t) {
        float z = Z3[((size_t)t * BB + b) * 2 + o];
        cur = cur * 0.7f + z;
        vol = vol * 0.2f + cur;
        float s = (vol >= 1.0f) ? 1.0f : 0.0f;
        vol *= (1.0f - s);
        out[((size_t)b * 2 + o) * TT + t] = s;
    }
}

// ---------------------------------------------------------------------------
extern "C" void kernel_launch(void* const* d_in, const int* in_sizes, int n_in,
                              void* d_out, int out_size, void* d_ws, size_t ws_size,
                              hipStream_t stream) {
    (void)in_sizes; (void)n_in; (void)out_size; (void)ws_size;
    const float* spike = (const float*)d_in[0];   // [B, NIN, T]
    const float* W1    = (const float*)d_in[1];   // [H, NIN]
    const float* W2    = (const float*)d_in[2];   // [H, H]
    const float* W3    = (const float*)d_in[3];   // [2, H]
    float* out = (float*)d_out;                   // [B, 2, T]

    // workspace carve-up (256B aligned), ~90 MB total
    char* ws = (char*)d_ws;
    _Float16* WH1 = (_Float16*)(ws + 0);                         //  4 MB
    _Float16* WH2 = (_Float16*)(ws + 4194304);                   //  2 MB
    _Float16* WH3 = (_Float16*)(ws + 6291456);                   //  4 KB
    _Float16* S0H = (_Float16*)(ws + 6295552);                   // 32 MB [NN,NIN]
    float*    Zb  = (float*)   (ws + 39849984);                  // 32 MB [NN,H]
    _Float16* S1H = (_Float16*)(ws + 73404416);                  // 16 MB [NN,H]
    _Float16* S2H = S0H;    // S0H is dead after GEMM1; reuse    // 16 MB [NN,H]
    float*    Z3  = (float*)   (ws + 90181632);                  // 64 KB [NN,2]

    // 1) weight converts
    cvt_f32_to_f16<<<(HH * NIN) / 256, 256, 0, stream>>>(W1, WH1, HH * NIN);
    cvt_f32_to_f16<<<(HH * HH) / 256, 256, 0, stream>>>(W2, WH2, HH * HH);
    cvt_f32_to_f16<<<(NOUT * HH) / 256, 256, 0, stream>>>(W3, WH3, NOUT * HH);

    // 2) spike transpose to [T*B, NIN] fp16
    spike_transpose<<<dim3(TT / 32, NIN / 32, BB), dim3(32, 8), 0, stream>>>(spike, S0H);

    // 3) layer 1 GEMM + scan
    wmma_gemm<<<dim3(HH / 64, NN / 128), 256, 0, stream>>>(WH1, S0H, Zb, HH, NIN);
    cuba_scan<<<(BB * HH) / 256, 256, 0, stream>>>(Zb, S1H, BB * HH);

    // 4) layer 2 GEMM + scan
    wmma_gemm<<<dim3(HH / 64, NN / 128), 256, 0, stream>>>(WH2, S1H, Zb, HH, HH);
    cuba_scan<<<(BB * HH) / 256, 256, 0, stream>>>(Zb, S2H, BB * HH);

    // 5) layer 3 + final scan
    layer3_z<<<NN / 8, 256, 0, stream>>>(WH3, S2H, Z3);
    final_scan<<<1, BB * NOUT, 0, stream>>>(Z3, out);
}
